// Lca_29892972380840
// MI455X (gfx1250) — matrix operand order repaired and checked by reading, not compiled, not run on hardware
//
#include <hip/hip_runtime.h>
#include <hip/hip_bf16.h>

#ifndef __has_builtin
#define __has_builtin(x) 0
#endif

#if defined(__AMDGCN__) && __has_builtin(__builtin_amdgcn_tensor_load_to_lds) && \
    __has_builtin(__builtin_amdgcn_s_wait_tensorcnt)
#define USE_TDM 1
#else
#define USE_TDM 0
#endif

#if defined(__AMDGCN__) && __has_builtin(__builtin_amdgcn_global_store_async_from_lds_b128)
#define USE_ASYNC_ST 1
#else
#define USE_ASYNC_ST 0
#endif

// ---------------- problem constants ----------------
#define BATCH   64
#define LSIG    65536
#define CCH     256
#define KF      1024
#define STR     128
#define S_LEN   505          // (65536-1024)/128 + 1
#define NK      3840         // 256 ch * 15 offsets
#define KSTEPS  120          // NK/32
#define NTILES  8            // 512/64 : 64-position tiles per batch
#define CTILES  16
#define BST     512          // padded pos stride for b,u (f32)
#define APOSS   528          // a layout: [batch][posSlot 0..527][ch 0..255] bf16
#define APAD    7            // pos p stored at slot p+7
#define LDST    264          // LDS row stride in halves (512B data + 16B pad)
#define LROWS   80           // pos-slots staged per WG (64 tile + 15 halo + round)
#define ETA     0.1f
#define THR     0.1f
#define ITERS   50

typedef __bf16 bf16_t;
typedef __attribute__((ext_vector_type(16))) __bf16 v16bf;
typedef __attribute__((ext_vector_type(8)))  float  v8f;
typedef unsigned int u32x4 __attribute__((ext_vector_type(4)));
typedef int          i32x4 __attribute__((ext_vector_type(4)));
typedef int          i32x8 __attribute__((ext_vector_type(8)));
typedef int          v4i_t __attribute__((ext_vector_type(4)));

__device__ __forceinline__ bf16_t f2bf(float f) {
    unsigned int x = __builtin_bit_cast(unsigned int, f);
    unsigned int r = (x + 0x7fffu + ((x >> 16) & 1u)) >> 16;
    unsigned short hs = (unsigned short)r;
    return __builtin_bit_cast(bf16_t, hs);
}

__device__ __forceinline__ unsigned lds_off_of(const void* p) {
#if defined(__AMDGCN__)
    return (unsigned)(uintptr_t)(__attribute__((address_space(3))) const void*)p;
#else
    return 0u;
#endif
}

// A-fragment (16x32 bf16) k-index for (lane, element e) per CDNA5 ISA layout.
__device__ __forceinline__ int a_frag_k(int lane, int e) {
    int basek = (lane < 16) ? 0 : 8;
    int v = e >> 1, h = e & 1;
    return (v < 4) ? (basek + 2 * v + h) : (basek + 16 + 2 * (v - 4) + h);
}

// ---------------------------------------------------------------------------
// Kernel 1: b = conv1d(x, W, stride=128)   (one-time, f32)
// ---------------------------------------------------------------------------
__global__ __launch_bounds__(256)
void b_conv(const float* __restrict__ x, const float* __restrict__ W,
            float* __restrict__ bout)
{
    __shared__ float xs[2944];
    const int batch = blockIdx.y;
    const int s0    = blockIdx.x * 16;
    const int base  = s0 * STR;

    for (int i = threadIdx.x; i < 2944; i += 256) {
        int gi = base + i;
        xs[i] = (gi < LSIG) ? x[(size_t)batch * LSIG + gi] : 0.0f;
    }
    __syncthreads();

    const int sl = threadIdx.x >> 4;
    const int cw = threadIdx.x & 15;
    const int s  = s0 + sl;
    const float* xr = xs + sl * STR;

    for (int cg = 0; cg < 16; ++cg) {
        int c = cg * 16 + cw;
        const float* Wr = W + (size_t)c * KF;
        float sum = 0.f;
        #pragma unroll 8
        for (int k = 0; k < KF; ++k) sum += xr[k] * Wr[k];
        if (s < S_LEN) bout[((size_t)(batch * CCH + c)) * BST + s] = sum;
    }
}

// ---------------------------------------------------------------------------
// Kernel 2: Gram tensor in WMMA A-fragment layout (bf16).
// ---------------------------------------------------------------------------
__global__ __launch_bounds__(256)
void gram_build(const float* __restrict__ W, bf16_t* __restrict__ Gf)
{
    size_t gid = (size_t)blockIdx.x * 256 + threadIdx.x;   // 0 .. 983040
    int e    = (int)(gid & 15);
    int lane = (int)((gid >> 4) & 31);
    int ks   = (int)((gid >> 9) % KSTEPS);
    int tile = (int)(gid / (16u * 32u * KSTEPS));

    int row = lane & 15;
    int k   = ks * 32 + a_frag_k(lane, e);
    int c   = k & 255;
    int d   = k >> 8;
    int cp  = tile * 16 + row;
    int off = (d - 7) * STR;
    int t0  = (off < 0) ? -off : 0;
    int t1  = (off > 0) ? (KF - off) : KF;

    const float* w1 = W + (size_t)cp * KF;
    const float* w2 = W + (size_t)c  * KF + off;
    float sum = 0.f;
    for (int t = t0; t < t1; ++t) sum += w1[t] * w2[t];
    Gf[gid] = f2bf(sum);
}

// ---------------------------------------------------------------------------
// Kernel 3: u0 = eta*b ; a0 = hardshrink(u0) into transposed a layout
// ---------------------------------------------------------------------------
__global__ __launch_bounds__(256)
void init_ua(const float* __restrict__ bin, float* __restrict__ u,
             bf16_t* __restrict__ a0)
{
    size_t gid = (size_t)blockIdx.x * 256 + threadIdx.x;   // over [b][c][pos]
    int pos    = (int)(gid & (BST - 1));
    int c      = (int)((gid >> 9) & 255);
    int batch  = (int)(gid >> 17);
    if (pos < S_LEN) {
        float uv = ETA * bin[gid];
        u[gid] = uv;
        float av = (fabsf(uv) > THR) ? uv : 0.0f;
        a0[((size_t)(batch * APOSS + pos + APAD)) * CCH + c] = f2bf(av);
    }
}

// ---------------------------------------------------------------------------
// Kernel 4: fused LCA step.  TDM-staged a-patch, WMMA core (4x N-blocking),
// LDS-transposed epilogue, async-store drain.
// ---------------------------------------------------------------------------
__global__ __launch_bounds__(256)
void lca_step(const bf16_t* __restrict__ Gf,
              const float*  __restrict__ bin,
              float*        __restrict__ u,
              const bf16_t* __restrict__ a_cur,
              bf16_t*       __restrict__ a_nxt,
              float*        __restrict__ outp,
              int write_out)
{
    __shared__ unsigned short ls[LROWS * LDST]; // [posSlot 0..79][ch 0..255] (+pad)

    const int batch = blockIdx.y;
    const int s0    = blockIdx.x * 64;
    const int tid   = threadIdx.x;
    const int lane  = tid & 31;
    const int wave  = tid >> 5;

    // ---- stage a-patch: rows (posSlot) s0 .. s0+79, all 256 ch, row-padded ----
    const char* gsrc = (const char*)(a_cur + (size_t)(batch * APOSS + s0) * CCH);
#if USE_TDM
    if (tid == 0) {
        unsigned long long ga = (unsigned long long)(uintptr_t)gsrc;
        u32x4 g0;
        g0[0] = 1u;                                   // count=1 (user descriptor)
        g0[1] = lds_off_of(ls);                       // lds_addr
        g0[2] = (unsigned)(ga & 0xffffffffu);         // global_addr[31:0]
        g0[3] = (unsigned)((ga >> 32) & 0x1ffffffu)   // global_addr[56:32]
              | (2u << 30);                           // type=2 (image)
        i32x8 g1;
        g1[0] = (int)((2u << 16)      // data_size = 4B
                    | (1u << 20)      // pad_enable
                    | (6u << 22)      // pad_interval: 128 dwords (one 512B row)
                    | (3u << 25));    // pad_amount : 4 dwords (16B)
        g1[1] = (int)(128u << 16);    // tensor_dim0 = 128 dwords
        g1[2] = (int)(80u << 16);     // tensor_dim0 hi=0 | tensor_dim1 = 80 rows
        g1[3] = (int)(128u << 16);    // tensor_dim1 hi=0 | tile_dim0 = 128
        g1[4] = 80;                   // tile_dim1 = 80 | tile_dim2 = 0
        g1[5] = 128;                  // tensor_dim0_stride = 128 dwords
        g1[6] = 0;
        g1[7] = 0;
        i32x4 z4 = {0, 0, 0, 0};
        i32x8 z8 = {0, 0, 0, 0, 0, 0, 0, 0};
        __builtin_amdgcn_tensor_load_to_lds(g0, g1, z4, z4, z8, 0);
        __builtin_amdgcn_s_wait_tensorcnt(0);
    }
#else
    {
        #pragma unroll
        for (int q = 0; q < 10; ++q) {
            int i   = tid + 256 * q;          // 2560 16B chunks
            int row = i >> 5;
            int off = (i & 31) * 16;
            int4 v = *(const int4*)(gsrc + (size_t)row * 512 + off);
            *(int4*)((char*)ls + row * 528 + off) = v;
        }
    }
#endif
    __syncthreads();

    const int col    = lane & 15;
    const int kbaseB = (lane < 16) ? 0 : 8;
    const int tile0  = wave;
    const int tile1  = wave + 8;

    v8f acc[2][4] = {{{}, {}, {}, {}}, {{}, {}, {}, {}}};   // [c'-tile][n-tile]
    const bf16_t* Ga0 = Gf + ((size_t)tile0 * KSTEPS * 32 + lane) * 16;
    const bf16_t* Ga1 = Gf + ((size_t)tile1 * KSTEPS * 32 + lane) * 16;

    #pragma unroll 2
    for (int ks = 0; ks < KSTEPS; ++ks) {
        const int kk = ks * 32;
        const int d  = kk >> 8;
        const int cc = (kk & 255) + kbaseB;

        union { int4 i[2]; v16bf v; } A0, A1;
        const int4* ap0 = (const int4*)(Ga0 + (size_t)ks * 32 * 16);
        const int4* ap1 = (const int4*)(Ga1 + (size_t)ks * 32 * 16);
        A0.i[0] = ap0[0]; A0.i[1] = ap0[1];
        A1.i[0] = ap1[0]; A1.i[1] = ap1[1];

        __builtin_prefetch(Ga0 + (size_t)(ks + 4) * 32 * 16, 0, 1);
        __builtin_prefetch(Ga1 + (size_t)(ks + 4) * 32 * 16, 0, 1);

        #pragma unroll
        for (int nt = 0; nt < 4; ++nt) {
            union { int4 i[2]; v16bf v; } B;
            const int4* bp = (const int4*)&ls[(col + 16 * nt + d) * LDST + cc];
            B.i[0] = bp[0]; B.i[1] = bp[2];
            acc[0][nt] = __builtin_amdgcn_wmma_f32_16x16x32_bf16(
                             false, A0.v, false, B.v, (short)0, acc[0][nt], false, false);
            acc[1][nt] = __builtin_amdgcn_wmma_f32_16x16x32_bf16(
                             false, A1.v, false, B.v, (short)0, acc[1][nt], false, false);
        }
    }

    // ---- epilogue: u update + hardshrink; a-tile transposed through LDS ----
    __syncthreads();                        // all B-fragment reads done; reuse ls

    const int n    = lane & 15;
    const int moff = (lane < 16) ? 0 : 8;
    #pragma unroll
    for (int nt = 0; nt < 4; ++nt) {
        const int pos = s0 + nt * 16 + n;
        if (pos < S_LEN) {
            #pragma unroll
            for (int t = 0; t < 2; ++t) {
                int ct = t ? tile1 : tile0;
                #pragma unroll
                for (int v = 0; v < 8; ++v) {
                    int c = ct * 16 + moff + v;
                    size_t idx = ((size_t)(batch * CCH + c)) * BST + pos;
                    float uo = u[idx];
                    float ao = (fabsf(uo) > THR) ? uo : 0.0f;
                    float un = uo + ETA * (bin[idx] - uo - acc[t][nt][v] + ao);
                    float an = (fabsf(un) > THR) ? un : 0.0f;
                    u[idx] = un;
                    ls[(nt * 16 + n) * LDST + c] =
                        __builtin_bit_cast(unsigned short, f2bf(an));
                    if (write_out)
                        outp[((size_t)(batch * CCH + c)) * S_LEN + pos] = an;
                }
            }
        }
    }
    __syncthreads();

    // ---- drain 64 rows x 512B to a_nxt (contiguous rows) ----
    char* gdst = (char*)(a_nxt + (size_t)(batch * APOSS + s0 + APAD) * CCH);
#if USE_ASYNC_ST
    #pragma unroll
    for (int q = 0; q < 8; ++q) {
        int j   = tid + 256 * q;            // 2048 16B chunks
        int r   = j >> 5;
        int off = (j & 31) * 16;
        if (s0 + r < S_LEN) {
            __attribute__((address_space(1))) v4i_t* gp =
                (__attribute__((address_space(1))) v4i_t*)(gdst + (size_t)r * 512 + off);
            __attribute__((address_space(3))) v4i_t* lp =
                (__attribute__((address_space(3))) v4i_t*)((char*)ls + r * 528 + off);
            __builtin_amdgcn_global_store_async_from_lds_b128(gp, lp, 0, 0);
        }
    }
    // S_ENDPGM performs an implicit wait-idle (covers ASYNCcnt) before the
    // next kernel in the stream can observe a_nxt.
#else
    {
        const unsigned int* ls32 = (const unsigned int*)ls;
        unsigned int* g32 = (unsigned int*)gdst;
        #pragma unroll
        for (int q = 0; q < 32; ++q) {
            int j = tid + 256 * q;          // 8192 dwords
            int r = j >> 7;
            int w = j & 127;
            if (s0 + r < S_LEN) g32[(size_t)r * 128 + w] = ls32[r * 132 + w];
        }
    }
#endif
}

// ---------------------------------------------------------------------------
// host launcher
// ---------------------------------------------------------------------------
extern "C" void kernel_launch(void* const* d_in, const int* in_sizes, int n_in,
                              void* d_out, int out_size, void* d_ws, size_t ws_size,
                              hipStream_t stream)
{
    const float* x = (const float*)d_in[0];   // [64,1,65536]
    const float* W = (const float*)d_in[1];   // [256,1,1024]
    float* outp = (float*)d_out;              // [64,256,505]

    const size_t gf_bytes = (size_t)CTILES * KSTEPS * 32 * 16 * sizeof(bf16_t); // 1,966,080
    const size_t bu_bytes = (size_t)BATCH * CCH * BST * sizeof(float);          // 33,554,432
    const size_t a_bytes  = (size_t)BATCH * APOSS * CCH * sizeof(bf16_t);       // 17,301,504

    char* ws = (char*)d_ws;
    bf16_t* Gf  = (bf16_t*)(ws);
    float*  bin = (float*)(ws + gf_bytes);
    float*  u   = (float*)(ws + gf_bytes + bu_bytes);
    bf16_t* a0  = (bf16_t*)(ws + gf_bytes + 2 * bu_bytes);
    bf16_t* a1  = (bf16_t*)(ws + gf_bytes + 2 * bu_bytes + a_bytes);

    (void)hipMemsetAsync(a0, 0, a_bytes, stream);
    (void)hipMemsetAsync(a1, 0, a_bytes, stream);

    b_conv<<<dim3(32, BATCH), 256, 0, stream>>>(x, W, bin);
    gram_build<<<dim3((CTILES * KSTEPS * 32 * 16) / 256), 256, 0, stream>>>(W, Gf);
    init_ua<<<dim3((BATCH * CCH * BST) / 256), 256, 0, stream>>>(bin, u, a0);

    bf16_t* cur = a0;
    bf16_t* nxt = a1;
    for (int it = 0; it < ITERS; ++it) {
        lca_step<<<dim3(NTILES, BATCH), 256, 0, stream>>>(
            Gf, bin, u, cur, nxt, outp, (it == ITERS - 1) ? 1 : 0);
        bf16_t* tmp = cur; cur = nxt; nxt = tmp;
    }
}
